// RGAT_5583457485246
// MI455X (gfx1250) — matrix operand order repaired and checked by reading, not compiled, not run on hardware
//
#include <hip/hip_runtime.h>

#define NFEAT 128
#define NHEAD 4

typedef __attribute__((ext_vector_type(16))) __bf16 v16bf;
typedef __attribute__((ext_vector_type(8)))  float  v8f;

union FragAB { unsigned u[8]; uint4 q[2]; v16bf v; };
union FragC  { float    f[8]; v8f   v; };

// ---------- helpers ----------
__device__ __forceinline__ unsigned f2bf(float f) {
  unsigned x = __float_as_uint(f);
  x += 0x7fffu + ((x >> 16) & 1u);           // round-to-nearest-even
  return x >> 16;
}
__device__ __forceinline__ unsigned pack2bf(float a, float b) {
  return f2bf(a) | (f2bf(b) << 16);
}
// order-preserving float<->int key for atomicMax-based segment max
__device__ __forceinline__ int fkey(float f) {
  int i = __float_as_int(f);
  return i >= 0 ? i : (i ^ 0x7fffffff);
}
__device__ __forceinline__ float fdec(int k) {
  if (k == (int)0x80000000) return 0.0f;     // sentinel: no incoming edges -> m=0
  return __int_as_float(k >= 0 ? k : (k ^ 0x7fffffff));
}

// ---------- generic fill ----------
__global__ void fill_u32_kernel(unsigned* __restrict__ p, unsigned v, int n) {
  int i = blockIdx.x * 256 + threadIdx.x;
  if (i < n) p[i] = v;
}

// ---------- pack fp32 weight (K x fout, row-major) into bf16 WMMA B-fragment layout ----------
// Lane-major so a lane's 8-dword fragment is contiguous (-> 2x global_load_b128):
// Wpk[(((ct*4 + ks)*32) + lane)*8 + v] : lane&15 = N, lane>>4 = K-half, 2 bf16/dword
__global__ void pack_w_kernel(const float* __restrict__ W, unsigned* __restrict__ Wpk,
                              int ncoltiles, int fout) {
  int idx = blockIdx.x * 256 + threadIdx.x;
  if (idx >= ncoltiles * 1024) return;
  int v = idx & 7, lane = (idx >> 3) & 31, ks = (idx >> 8) & 3, ct = idx >> 10;
  int n  = ct * 16 + (lane & 15);
  int k0 = ks * 32 + (lane >> 4) * 16 + 2 * v;
  Wpk[idx] = pack2bf(W[(size_t)k0 * fout + n], W[(size_t)(k0 + 1) * fout + n]);
}

// ---------- WMMA GEMM: Y[rows x fout] = X[rows x 128] @ W + bias ----------
// one block = one 16-row strip; 8 waves, wave w computes column tile w (16 cols)
__global__ __launch_bounds__(256)
void gemm_wmma_kernel(const float* __restrict__ X, const unsigned* __restrict__ Wpk,
                      const float* __restrict__ bias, float* __restrict__ Y,
                      int ncoltiles, int ldY) {
  __shared__ unsigned ldsA[16 * 68];               // 16 rows x 64 packed-bf16 dwords, pad 68
  const int tid  = threadIdx.x;
  const int row0 = blockIdx.x * 16;

  // cooperative load: 16x128 fp32 -> packed bf16 in LDS (coalesced float2 reads)
  #pragma unroll
  for (int i = 0; i < 4; ++i) {
    int idx = i * 256 + tid;                       // 0..1023
    int r = idx >> 6, c = idx & 63;
    const float2 xv = *(const float2*)(X + (size_t)(row0 + r) * NFEAT + 2 * c);
    ldsA[r * 68 + c] = pack2bf(xv.x, xv.y);
  }
  __syncthreads();

  const int wave = tid >> 5, lane = tid & 31;
  if (wave >= ncoltiles) return;                   // wave-uniform

  const int ct = wave;
  const int m = lane & 15, g = lane >> 4;

  FragC c;
  #pragma unroll
  for (int i = 0; i < 8; ++i) c.f[i] = 0.0f;

  #pragma unroll
  for (int ks = 0; ks < 4; ++ks) {
    FragAB a, b;
    // A 16x32 bf16 in this code's contiguous k order (matches Wpk): lanes0-15 row m, K=2v (v=0..7); lanes16-31 K=16+2v
    const int base = m * 68 + ks * 16 + g * 8;
    #pragma unroll
    for (int v = 0; v < 4; ++v) {
      a.u[v]     = ldsA[base + v];
      a.u[v + 4] = ldsA[base + 4 + v];
    }
    // B fragment: contiguous 8 dwords per lane -> two b128 loads, wave reads 1KB slab
    const uint4* wp = (const uint4*)(Wpk + ((size_t)((ct * 4 + ks) * 32) + lane) * 8);
    b.q[0] = wp[0];
    b.q[1] = wp[1];
    c.v = __builtin_amdgcn_wmma_f32_16x16x32_bf16(false, a.v, false, b.v,
                                                  (short)0, c.v, false, false);
  }

  const int n = ct * 16 + m;
  const float bval = bias ? bias[n] : 0.0f;
  #pragma unroll
  for (int r = 0; r < 8; ++r) {
    int row = row0 + r + 8 * g;                    // C/D layout: VGPR r -> M = r + 8*g
    Y[(size_t)row * ldY + n] = c.f[r] + bval;
  }
}

// ---------- el/er = einsum('nhd,hd->nh') ----------
__global__ void att_dot_kernel(const float* __restrict__ Z, const float* __restrict__ avec,
                               float* __restrict__ out, int n_nodes) {
  int idx = blockIdx.x * 256 + threadIdx.x;
  if (idx >= n_nodes * NHEAD) return;
  int n = idx >> 2, h = idx & 3;
  const float* z = Z + (size_t)n * NFEAT + h * 32;
  const float* a = avec + h * 32;
  float s = 0.0f;
  #pragma unroll
  for (int d = 0; d < 32; ++d) s += z[d] * a[d];
  out[idx] = s;
}

// ---------- edge pass 1: segment max of leaky_relu(el[s]+er[d]) ----------
__global__ void edge_max_kernel(const int* __restrict__ src, const int* __restrict__ dst,
                                const float* __restrict__ el, const float* __restrict__ er,
                                int* __restrict__ mkey, int E) {
  int idx = blockIdx.x * 256 + threadIdx.x;
  if (idx >= E * NHEAD) return;
  int e = idx >> 2, h = idx & 3;
  int s = src[e], d = dst[e];
  float x = el[s * NHEAD + h] + er[d * NHEAD + h];
  x = x > 0.0f ? x : 0.2f * x;
  atomicMax(&mkey[d * NHEAD + h], fkey(x));
}

// ---------- edge pass 2: den = segment sum of exp(e - m[d]) ----------
__global__ void edge_expsum_kernel(const int* __restrict__ src, const int* __restrict__ dst,
                                   const float* __restrict__ el, const float* __restrict__ er,
                                   const int* __restrict__ mkey, float* __restrict__ den, int E) {
  int idx = blockIdx.x * 256 + threadIdx.x;
  if (idx >= E * NHEAD) return;
  int e = idx >> 2, h = idx & 3;
  int s = src[e], d = dst[e];
  float x = el[s * NHEAD + h] + er[d * NHEAD + h];
  x = x > 0.0f ? x : 0.2f * x;
  float m = fdec(mkey[d * NHEAD + h]);
  atomicAdd(&den[d * NHEAD + h], __expf(x - m));
}

// ---------- edge pass 3: acc[d] += alpha * z_src[s] (wave32 per edge, 4 floats/lane) ----------
__global__ __launch_bounds__(256)
void edge_scatter_kernel(const int* __restrict__ src, const int* __restrict__ dst,
                         const float* __restrict__ el, const float* __restrict__ er,
                         const int* __restrict__ mkey, const float* __restrict__ den,
                         const float* __restrict__ zS, float* __restrict__ acc, int E) {
  int e = blockIdx.x * 8 + (threadIdx.x >> 5);
  if (e >= E) return;                              // wave-uniform
  int lane = threadIdx.x & 31;
  int s = src[e], d = dst[e];
  int h = lane >> 3;
  float x = el[s * NHEAD + h] + er[d * NHEAD + h];
  x = x > 0.0f ? x : 0.2f * x;
  float m  = fdec(mkey[d * NHEAD + h]);
  float ex = __expf(x - m);
  float dn = den[d * NHEAD + h];
  float alpha = ex / (dn > 0.0f ? dn : 1.0f);
  int d0 = (lane & 7) * 4;
  const float4 zv = *(const float4*)(zS + (size_t)s * NFEAT + h * 32 + d0);
  float* o = acc + (size_t)d * NFEAT + h * 32 + d0;
  atomicAdd(o + 0, alpha * zv.x);
  atomicAdd(o + 1, alpha * zv.y);
  atomicAdd(o + 2, alpha * zv.z);
  atomicAdd(o + 3, alpha * zv.w);
}

// ---------- combine: hp = act(0.5*(oW+oC) + 0.5*(bA+bB)) / ha = act(oWB + b) ----------
__global__ void combine2_kernel(const float* __restrict__ acc, const float* __restrict__ bA,
                                const float* __restrict__ bB, float* __restrict__ out,
                                int n, int do_elu) {
  int idx = blockIdx.x * 256 + threadIdx.x;
  if (idx >= n * NFEAT) return;
  int f = idx & (NFEAT - 1);
  float v = 0.5f * acc[idx] + 0.5f * (bA[f] + bB[f]);
  if (do_elu) v = v > 0.0f ? v : (__expf(v) - 1.0f);
  out[idx] = v;
}
__global__ void combine1_kernel(const float* __restrict__ acc, const float* __restrict__ b,
                                float* __restrict__ out, int n, int do_elu) {
  int idx = blockIdx.x * 256 + threadIdx.x;
  if (idx >= n * NFEAT) return;
  int f = idx & (NFEAT - 1);
  float v = acc[idx] + b[f];
  if (do_elu) v = v > 0.0f ? v : (__expf(v) - 1.0f);
  out[idx] = v;
}

// ---------- host ----------
extern "C" void kernel_launch(void* const* d_in, const int* in_sizes, int n_in,
                              void* d_out, int out_size, void* d_ws, size_t ws_size,
                              hipStream_t stream) {
  const float* x_p = (const float*)d_in[0];
  const float* x_a = (const float*)d_in[1];
  const int* srcs[3] = {(const int*)d_in[2], (const int*)d_in[4], (const int*)d_in[6]};
  const int* dsts[3] = {(const int*)d_in[3], (const int*)d_in[5], (const int*)d_in[7]};
  const int  Ecnt[3] = {in_sizes[2], in_sizes[4], in_sizes[6]};
  const float* Wl[2]  = {(const float*)d_in[8],  (const float*)d_in[12]};
  const float* alv[2] = {(const float*)d_in[9],  (const float*)d_in[13]};
  const float* arv[2] = {(const float*)d_in[10], (const float*)d_in[14]};
  const float* bl[2]  = {(const float*)d_in[11], (const float*)d_in[15]};
  const float* Wc = (const float*)d_in[16];
  const float* bc = (const float*)d_in[17];

  const int N  = in_sizes[0] / NFEAT;   // 50000, divisible by 16
  const int NH = N * NHEAD;
  (void)n_in; (void)out_size; (void)ws_size;

  // carve workspace
  char* wsb = (char*)d_ws; size_t off = 0;
  auto carve = [&](size_t bytes) -> void* {
    void* p = wsb + off; off += (bytes + 255) & ~(size_t)255; return p;
  };
  float* zS   = (float*)carve((size_t)N * NFEAT * 4);
  float* zD   = (float*)carve((size_t)N * NFEAT * 4);
  float* elb  = (float*)carve((size_t)NH * 4);
  float* erb  = (float*)carve((size_t)NH * 4);
  int*   mkey = (int*)  carve((size_t)NH * 4);
  float* den  = (float*)carve((size_t)NH * 4);
  float* accP = (float*)carve((size_t)N * NFEAT * 4);
  float* accA = (float*)carve((size_t)N * NFEAT * 4);
  float* hp   = (float*)carve((size_t)N * NFEAT * 4);
  float* ha   = (float*)carve((size_t)N * NFEAT * 4);
  unsigned* wpk[6];
  for (int i = 0; i < 6; ++i) wpk[i] = (unsigned*)carve(8192 * 4);
  unsigned* wpkc = (unsigned*)carve(1024 * 4);

  // pack all weights into WMMA bf16 fragment layout
  for (int L = 0; L < 2; ++L)
    for (int r = 0; r < 3; ++r)
      pack_w_kernel<<<32, 256, 0, stream>>>(Wl[L] + (size_t)r * NFEAT * NFEAT,
                                            wpk[L * 3 + r], 8, NFEAT);
  pack_w_kernel<<<4, 256, 0, stream>>>(Wc, wpkc, 1, 16);

  const int rows16 = N / 16;
  const int nfTot  = N * NFEAT;
  const int gNF    = (nfTot + 255) / 256;
  const int gNH    = (NH + 255) / 256;
  const float* hpin = x_p;
  const float* hain = x_a;

  for (int L = 0; L < 2; ++L) {
    fill_u32_kernel<<<gNF, 256, 0, stream>>>((unsigned*)accP, 0u, nfTot);
    if (L == 0)
      fill_u32_kernel<<<gNF, 256, 0, stream>>>((unsigned*)accA, 0u, nfTot);

    const int nrel = (L == 0) ? 3 : 2;   // layer-2 author output never reaches classifier
    for (int r = 0; r < nrel; ++r) {
      const float* srcFeat = (r == 0) ? hain : hpin;
      const float* dstFeat = (r == 2) ? hain : hpin;
      float*       accT    = (r == 2) ? accA : accP;

      gemm_wmma_kernel<<<rows16, 256, 0, stream>>>(srcFeat, wpk[L * 3 + r], nullptr, zS, 8, NFEAT);
      const float* zDp = zS;               // cites: src==dst projection shared
      if (r != 1) {
        gemm_wmma_kernel<<<rows16, 256, 0, stream>>>(dstFeat, wpk[L * 3 + r], nullptr, zD, 8, NFEAT);
        zDp = zD;
      }
      att_dot_kernel<<<gNH, 256, 0, stream>>>(zS,  alv[L] + r * NFEAT, elb, N);
      att_dot_kernel<<<gNH, 256, 0, stream>>>(zDp, arv[L] + r * NFEAT, erb, N);
      fill_u32_kernel<<<gNH, 256, 0, stream>>>((unsigned*)mkey, 0x80000000u, NH);
      fill_u32_kernel<<<gNH, 256, 0, stream>>>((unsigned*)den, 0u, NH);

      const int E = Ecnt[r];
      edge_max_kernel   <<<(E * 4 + 255) / 256, 256, 0, stream>>>(srcs[r], dsts[r], elb, erb, mkey, E);
      edge_expsum_kernel<<<(E * 4 + 255) / 256, 256, 0, stream>>>(srcs[r], dsts[r], elb, erb, mkey, den, E);
      edge_scatter_kernel<<<(E + 7) / 8, 256, 0, stream>>>(srcs[r], dsts[r], elb, erb, mkey, den, zS, accT, E);
    }

    if (L == 0) {
      combine2_kernel<<<gNF, 256, 0, stream>>>(accP, bl[0], bl[0] + NFEAT, hp, N, 1);
      combine1_kernel<<<gNF, 256, 0, stream>>>(accA, bl[0] + 2 * NFEAT, ha, N, 1);
      hpin = hp; hain = ha;
    } else {
      combine2_kernel<<<gNF, 256, 0, stream>>>(accP, bl[1], bl[1] + NFEAT, hp, N, 0);
    }
  }

  // classifier: d_out = hp2 @ Wc + bc  (50000 x 16)
  gemm_wmma_kernel<<<rows16, 256, 0, stream>>>(hp, wpkc, bc, (float*)d_out, 1, 16);
}
